// Processor_34832184771138
// MI455X (gfx1250) — matrix operand order repaired and checked by reading, not compiled, not run on hardware
//
#include <hip/hip_runtime.h>
#include <stdint.h>

// ---------------------------------------------------------------------------
// CGConv on MI455X (gfx1250): bf16 WMMA edge-GEMM + f32 atomic scatter-add.
// 64-edge tiles: each wave owns 16 output channels x 4 edge M-tiles, so each
// B (weight) fragment is reused by 4 WMMAs per matrix -> load-issue balanced.
// ---------------------------------------------------------------------------

typedef __attribute__((ext_vector_type(16))) __bf16 v16bf;
typedef __attribute__((ext_vector_type(8)))  float  v8f;
typedef __attribute__((ext_vector_type(4)))  float  f32x4;   // native vec for NT loads

#define NCH       128          // node channels C
#define ZDIM      384          // 2C + D
#define TILE_E    64           // edges per block
#define MTILES    4            // 64 / 16 M-tiles per wave
#define KSTEPS    12           // 384 / 32
#define NTILES    8            // 128 / 16
#define ROW_PITCH 392          // 384 halves + 8 halves pad (784 B, 16B aligned)

struct Q2 { uint4 a, b; };     // 32 bytes == v16bf

__device__ __forceinline__ unsigned short f2bf(float f) {
    unsigned u = __float_as_uint(f);
    unsigned r = u + 0x7FFFu + ((u >> 16) & 1u);   // round-to-nearest-even
    return (unsigned short)(r >> 16);
}
__device__ __forceinline__ unsigned pack2(float a, float b) {
    return (unsigned)f2bf(a) | ((unsigned)f2bf(b) << 16);
}

// ---------------------------------------------------------------------------
// Prep 1: out = x (fp32 copy) and x_bf16 = bf16(x) into workspace.
// ---------------------------------------------------------------------------
__global__ void prep_x_kernel(const float* __restrict__ x,
                              float* __restrict__ out,
                              unsigned* __restrict__ xbf, int n2) {
    int i = blockIdx.x * blockDim.x + threadIdx.x;
    if (i >= n2) return;
    float2 v = ((const float2*)x)[i];
    ((float2*)out)[i] = v;
    xbf[i] = pack2(v.x, v.y);
}

// ---------------------------------------------------------------------------
// Prep 2: pack W_f / W_s (384x128 f32, row-major [K][N]) into the wave32
// bf16 B-fragment layout: frag[(ntile*12+kk)][lane][v] holds the K pair
// (K0, K0+1) for column = ntile*16 + lane%16, K0 = kk*32 + 16*(lane/16) + 2v.
// ---------------------------------------------------------------------------
__global__ void prep_w_kernel(const float* __restrict__ Wf,
                              const float* __restrict__ Ws,
                              unsigned* __restrict__ fragF,
                              unsigned* __restrict__ fragS) {
    const int PER = NTILES * KSTEPS * 32 * 8;      // 24576 dwords / matrix
    int id = blockIdx.x * blockDim.x + threadIdx.x;
    if (id >= 2 * PER) return;
    const float* W   = (id < PER) ? Wf    : Ws;
    unsigned*    dst = (id < PER) ? fragF : fragS;
    int t    = (id < PER) ? id : id - PER;
    int v    = t & 7;
    int lane = (t >> 3) & 31;
    int frag = t >> 8;                             // 0..95 == ntile*12 + kk
    int ntile = frag / KSTEPS;
    int kk    = frag % KSTEPS;
    int col = ntile * 16 + (lane & 15);
    int k0  = kk * 32 + ((lane >> 4) << 4) + 2 * v;
    dst[t] = pack2(W[k0 * NCH + col], W[(k0 + 1) * NCH + col]);
}

// ---------------------------------------------------------------------------
// Main: one block = 64 edges, 8 waves. Wave w owns 16 output channels and
// all 4 edge M-tiles; B fragments loaded once per K-step, reused 4x.
// ---------------------------------------------------------------------------
__global__ __launch_bounds__(256)
void cgconv_edge_kernel(const int* __restrict__ ei,      // [2,E] (src row 0, dst row 1)
                        const float* __restrict__ ea,    // [E,128]
                        const float* __restrict__ bF,
                        const float* __restrict__ bS,
                        const unsigned* __restrict__ xbf,  // [N][64] bf16 pairs
                        const uint4* __restrict__ fragF,   // [96][32][2] uint4
                        const uint4* __restrict__ fragS,
                        float* __restrict__ out, int E) {
    __shared__ __align__(16) unsigned short zT[TILE_E * ROW_PITCH];
    __shared__ int sDst[TILE_E];

    const int  tid = threadIdx.x;
    const long e0  = (long)blockIdx.x * TILE_E;

    if (tid == 0 && e0 + TILE_E < E)
        __builtin_prefetch(ea + (e0 + TILE_E) * NCH, 0, 0);

    if (tid < TILE_E) {
        long e = e0 + tid; if (e >= E) e = E - 1;
        sDst[tid] = ei[E + e];                      // dst node i
    }

    // Stage z = [x_dst | x_src | e_attr] as bf16, 16B chunks (48 per row).
    for (int c = tid; c < TILE_E * 48; c += 256) {
        int row = c / 48, seg = c % 48;
        long e = e0 + row; if (e >= E) e = E - 1;
        uint4 val;
        int off;
        if (seg < 32) {
            int node = (seg < 16) ? ei[E + e] : ei[e];   // dst first, then src
            val = ((const uint4*)(xbf + (long)node * 64))[seg & 15];
            off = (seg < 16) ? (seg * 8) : (NCH + (seg - 16) * 8);
        } else {
            int j = (seg - 32) * 8;
            const f32x4* p = (const f32x4*)(ea + e * NCH + j);
            f32x4 f0 = __builtin_nontemporal_load(p);        // streamed once
            f32x4 f1 = __builtin_nontemporal_load(p + 1);
            val.x = pack2(f0.x, f0.y); val.y = pack2(f0.z, f0.w);
            val.z = pack2(f1.x, f1.y); val.w = pack2(f1.z, f1.w);
            off = 2 * NCH + j;
        }
        *((uint4*)&zT[row * ROW_PITCH + off]) = val;
    }
    __syncthreads();

    const int lane  = tid & 31;
    const int wave  = tid >> 5;                    // ntile 0..7
    const int arow  = lane & 15;                   // A-matrix row M (within tile)
    const int ksel  = (lane >> 4);                 // K-half select

    v8f accF[MTILES];
    v8f accS[MTILES];
    #pragma unroll
    for (int mt = 0; mt < MTILES; ++mt) { accF[mt] = {}; accS[mt] = {}; }

    const uint4* pF = fragF + (wave * KSTEPS) * 64 + lane * 2;
    const uint4* pS = fragS + (wave * KSTEPS) * 64 + lane * 2;

    #pragma unroll 4
    for (int kk = 0; kk < KSTEPS; ++kk) {
        // B fragments: one per matrix per K-step, reused by all 4 M-tiles.
        Q2 qf; qf.a = pF[kk * 64 + 0]; qf.b = pF[kk * 64 + 1];
        Q2 qs; qs.a = pS[kk * 64 + 0]; qs.b = pS[kk * 64 + 1];
        v16bf bfr = __builtin_bit_cast(v16bf, qf);
        v16bf bsr = __builtin_bit_cast(v16bf, qs);

        int kb = kk * 32 + (ksel << 4);
        #pragma unroll
        for (int mt = 0; mt < MTILES; ++mt) {
            // A fragment: lanes 0-15 hold K kb..kb+15 contiguous; lanes 16-31
            // the +16 shifted half. Two ds_load_b128.
            const unsigned short* zRow = &zT[(mt * 16 + arow) * ROW_PITCH];
            Q2 qa;
            qa.a = *(const uint4*)(zRow + kb);
            qa.b = *(const uint4*)(zRow + kb + 8);
            v16bf a = __builtin_bit_cast(v16bf, qa);

            accF[mt] = __builtin_amdgcn_wmma_f32_16x16x32_bf16(
                           false, a, false, bfr, (short)0, accF[mt], false, false);
            accS[mt] = __builtin_amdgcn_wmma_f32_16x16x32_bf16(
                           false, a, false, bsr, (short)0, accS[mt], false, false);
        }
    }

    // Epilogue: bias, sigmoid * softplus, fp32 atomic scatter-add.
    const int col = wave * 16 + (lane & 15);       // C/D layout: N = lane%16
    const float biasF = bF[col];
    const float biasS = bS[col];
    const int mbase = ksel * 8;                    // M = r + 8*(lane/16)

    #pragma unroll
    for (int mt = 0; mt < MTILES; ++mt) {
        #pragma unroll
        for (int r = 0; r < 8; ++r) {
            int  m = mt * 16 + mbase + r;
            long e = e0 + m;
            if (e < E) {
                float g = accF[mt][r] + biasF;
                g = 1.0f / (1.0f + __expf(-g));
                float s = accS[mt][r] + biasS;
                s = (s > 15.0f) ? s : log1pf(__expf(s));
                atomicAdd(&out[(long)sDst[m] * NCH + col], g * s);
            }
        }
    }
}

// ---------------------------------------------------------------------------
// Launch
// ---------------------------------------------------------------------------
extern "C" void kernel_launch(void* const* d_in, const int* in_sizes, int n_in,
                              void* d_out, int out_size, void* d_ws, size_t ws_size,
                              hipStream_t stream) {
    const float* x    = (const float*)d_in[0];
    const int*   ei   = (const int*)  d_in[1];
    const float* ea   = (const float*)d_in[2];
    const float* W_f  = (const float*)d_in[3];
    const float* b_f  = (const float*)d_in[4];
    const float* W_s  = (const float*)d_in[5];
    const float* b_s  = (const float*)d_in[6];
    float* out = (float*)d_out;

    const int nx = in_sizes[0];          // N * 128
    const int E  = in_sizes[1] / 2;      // 800000

    // Workspace layout (all 16B aligned):
    //   [0)  x_bf16 : nx/2 dwords
    //   [A)  fragF  : 24576 dwords
    //   [B)  fragS  : 24576 dwords
    char* ws = (char*)d_ws;
    unsigned* xbf   = (unsigned*)ws;
    size_t xbfBytes = (size_t)(nx / 2) * 4;
    xbfBytes = (xbfBytes + 15) & ~(size_t)15;
    unsigned* wfF = (unsigned*)(ws + xbfBytes);
    unsigned* wfS = wfF + NTILES * KSTEPS * 32 * 8;

    // 1) out = x, x_bf16 = bf16(x)
    {
        int n2 = nx / 2;
        int grid = (n2 + 255) / 256;
        prep_x_kernel<<<grid, 256, 0, stream>>>(x, out, xbf, n2);
    }
    // 2) pack both weight matrices into B-fragment layout
    {
        int total = 2 * NTILES * KSTEPS * 32 * 8;
        prep_w_kernel<<<(total + 255) / 256, 256, 0, stream>>>(W_f, W_s, wfF, wfS);
    }
    // 3) edge GEMM + activation + scatter-add
    {
        int grid = (E + TILE_E - 1) / TILE_E;
        cgconv_edge_kernel<<<grid, 256, 0, stream>>>(
            ei, ea, b_f, b_s, xbf, (const uint4*)wfF, (const uint4*)wfS, out, E);
    }
}
